// TPCNO_2714419331690
// MI455X (gfx1250) — matrix operand order repaired and checked by reading, not compiled, not run on hardware
//
#include <hip/hip_runtime.h>

// ---------------- types ----------------
typedef _Float16 f16;
typedef _Float16 v16h __attribute__((ext_vector_type(16)));
typedef _Float16 v8h  __attribute__((ext_vector_type(8)));
typedef float    v8f  __attribute__((ext_vector_type(8)));

// ---------------- problem constants ----------------
constexpr int B_  = 2;
constexpr int N_  = 20000;
constexpr int E_  = 100000;
constexpr int D_  = 3;
constexpr int C_  = 128;
constexpr int K_  = 128;
constexpr int FC_ = 128;
constexpr int L_  = 3;
constexpr int RS  = 768;   // stacked reduction rows: [bc(128) | bs(128) | h(128) | g(384)]

constexpr int FWD_STEPS  = 5;    // 5*32 = 160 nodes per split-K chunk
constexpr int FWD_CHUNKS = 125;  // 125*160 = 20000

#define DEVFN static __device__ __forceinline__

// ---- CDNA5 async global->LDS path (guarded; falls back to sync copy) ----
#if defined(__has_builtin)
#  if __has_builtin(__builtin_amdgcn_global_load_async_to_lds_b128) && \
      __has_builtin(__builtin_amdgcn_s_wait_asynccnt)
#    define USE_ASYNC 1
#  endif
#endif
#ifndef USE_ASYNC
#  define USE_ASYNC 0
#endif

#if USE_ASYNC
// Diagnostic from the toolchain shows the builtin's params are
// int4 addrspace(1)* (printed "__device__") and int4 addrspace(3)*.
typedef int v4i __attribute__((vector_size(4 * sizeof(int))));
typedef __attribute__((address_space(1))) v4i g_v4i;
typedef __attribute__((address_space(3))) v4i l_v4i;
DEVFN void async_g2l_b128(const f16* gp, f16* lp) {
  __builtin_amdgcn_global_load_async_to_lds_b128((g_v4i*)gp, (l_v4i*)lp, 0, 0);
}
#endif

DEVFN v16h ld2x8(const f16* p0, const f16* p1) {
  v8h a = *(const v8h*)p0;
  v8h b = *(const v8h*)p1;
  v16h r;
#pragma unroll
  for (int i = 0; i < 8; ++i) { r[i] = a[i]; r[i + 8] = b[i]; }
  return r;
}

DEVFN v8f wmma16(v16h a, v16h b, v8f c) {
  // D(16x16,f32) = A(16x32,f16) * B(32x16,f16) + C
  return __builtin_amdgcn_wmma_f32_16x16x32_f16(false, a, false, b, (short)0, c,
                                                false, false);
}

DEVFN float gelu_exact(float v) {
  return 0.5f * v * (1.0f + erff(v * 0.70710678118654752f));
}

// ---------------- kernels ----------------

__global__ void k_zero(float* p, size_t n) {
  size_t i  = (size_t)blockIdx.x * blockDim.x + threadIdx.x;
  size_t st = (size_t)gridDim.x * blockDim.x;
  for (; i < n; i += st) p[i] = 0.0f;
}

// Fourier basis: bc/bs rows of the stacked B matrix + node-weighted copies for
// the forward transform (mode-major, node-contiguous for coalesced WMMA B tiles).
__global__ void k_basis(const float* __restrict__ nodes,
                        const float* __restrict__ nodw,
                        const float* __restrict__ modes,
                        const float* __restrict__ spL,
                        f16* __restrict__ bstack, f16* __restrict__ wb,
                        float* __restrict__ nw) {
  __shared__ float mm[K_ * D_];
  for (int i = threadIdx.x; i < K_ * D_; i += blockDim.x)
    mm[i] = modes[i] * spL[i % D_];
  __syncthreads();
  int b = blockIdx.y;
  int x = blockIdx.x * blockDim.x + threadIdx.x;
  if (x >= N_) return;
  size_t nx = (size_t)b * N_ + x;
  float p0 = nodes[nx * 3 + 0], p1 = nodes[nx * 3 + 1], p2 = nodes[nx * 3 + 2];
  float w = nodw[nx];
  nw[nx] = w;
  f16* brow = bstack + nx * RS;
  for (int k = 0; k < K_; ++k) {
    float ph = p0 * mm[k * 3 + 0] + p1 * mm[k * 3 + 1] + p2 * mm[k * 3 + 2];
    float c = cosf(ph), s = sinf(ph);
    brow[k]      = (f16)c;
    brow[K_ + k] = (f16)s;
    wb[((size_t)b * 2 * K_ + k) * N_ + x]      = (f16)(c * w);
    wb[((size_t)b * 2 * K_ + K_ + k) * N_ + x] = (f16)(s * w);
  }
}

// h = [x,t] @ fc0_w + fc0_b, emitted in three layouts.
__global__ void k_fc0(const float* __restrict__ xin, const float* __restrict__ tin,
                      const float* __restrict__ w, const float* __restrict__ bias,
                      float* __restrict__ hN, f16* __restrict__ hA,
                      f16* __restrict__ bstack) {
  __shared__ float sw[4 * C_];
  __shared__ float sb[C_];
  for (int i = threadIdx.x; i < 4 * C_; i += blockDim.x) sw[i] = w[i];
  for (int i = threadIdx.x; i < C_; i += blockDim.x) sb[i] = bias[i];
  __syncthreads();
  int b = blockIdx.y;
  int x = blockIdx.x * blockDim.x + threadIdx.x;
  if (x >= N_) return;
  size_t nx = (size_t)b * N_ + x;
  float a0 = xin[nx * 3 + 0], a1 = xin[nx * 3 + 1], a2 = xin[nx * 3 + 2];
  float a3 = tin[nx];
  float* hrow = hN + nx * C_;
  f16* brow = bstack + nx * RS + 2 * K_;
  for (int c = 0; c < C_; ++c) {
    float v = a0 * sw[c] + a1 * sw[C_ + c] + a2 * sw[2 * C_ + c] +
              a3 * sw[3 * C_ + c] + sb[c];
    hrow[c] = v;
    hA[((size_t)b * C_ + c) * N_ + x] = (f16)v;
    brow[c] = (f16)v;
  }
}

// Forward transform (split-K over nodes): xc += h*wbc, xsraw += h*wbs.
// The 256x32 wb chunk is shared by all 8 waves -> stage once per block into
// LDS via async global->LDS, double buffered; A tiles read from global each step.
__global__ void __launch_bounds__(256)
k_fwd(const f16* __restrict__ hA, const f16* __restrict__ wb,
      float* __restrict__ xc, float* __restrict__ xs) {
  __shared__ f16 Bbuf[2][2 * K_ * 32];  // 2 x 16 KB

  int b = blockIdx.y, chunk = blockIdx.x;
  int t = threadIdx.x;
  int lane = t & 31, wv = t >> 5;
  int hi = lane >> 4, lo = lane & 15;
  int c0 = wv * 16;
  int xbase = chunk * (FWD_STEPS * 32);

  const f16* hbase = hA + (size_t)b * C_ * N_;
  const f16* wbase = wb + (size_t)b * 2 * K_ * N_;

  // Stage one 32-node slab of all 256 wb rows: row t, 4 x 16B segments each.
  auto issueB = [&](int ks, int buf) {
#pragma unroll
    for (int j = 0; j < 4; ++j) {
      const f16* gp = wbase + (size_t)t * N_ + xbase + ks * 32 + j * 8;
      f16* lp = &Bbuf[buf][t * 32 + j * 8];
#if USE_ASYNC
      async_g2l_b128(gp, lp);
#else
      *(v8h*)lp = *(const v8h*)gp;
#endif
    }
  };

  v8f z8 = {0.f, 0.f, 0.f, 0.f, 0.f, 0.f, 0.f, 0.f};
  v8f accC[8], accS[8];
#pragma unroll
  for (int i = 0; i < 8; ++i) { accC[i] = z8; accS[i] = z8; }

  issueB(0, 0);

  for (int ks = 0; ks < FWD_STEPS; ++ks) {
    int buf = ks & 1;
    if (ks + 1 < FWD_STEPS) {
      issueB(ks + 1, buf ^ 1);
#if USE_ASYNC
      __builtin_amdgcn_s_wait_asynccnt(4);  // this buffer's 4 are done
#endif
    } else {
#if USE_ASYNC
      __builtin_amdgcn_s_wait_asynccnt(0);
#endif
    }
    __syncthreads();

    int xcol = xbase + ks * 32;
    const f16* arow = hbase + (size_t)(c0 + lo) * N_ + xcol;
    v16h a = ld2x8(arow + 8 * hi, arow + 16 + 8 * hi);  // A 16x32 lane layout
#pragma unroll
    for (int nt = 0; nt < 8; ++nt) {
      const f16* bc_p = &Bbuf[buf][(nt * 16 + lo) * 32 + 8 * hi];
      accC[nt] = wmma16(a, ld2x8(bc_p, bc_p + 16), accC[nt]);
      const f16* bs_p = &Bbuf[buf][(K_ + nt * 16 + lo) * 32 + 8 * hi];
      accS[nt] = wmma16(a, ld2x8(bs_p, bs_p + 16), accS[nt]);
    }
    __syncthreads();  // protect buf from next iteration's async overwrite
  }

  float* xcb = xc + (size_t)b * C_ * K_;
  float* xsb = xs + (size_t)b * C_ * K_;
#pragma unroll
  for (int nt = 0; nt < 8; ++nt) {
#pragma unroll
    for (int v = 0; v < 8; ++v) {
      int c = c0 + 8 * hi + v;
      int k = nt * 16 + lo;
      atomicAdd(&xcb[(size_t)c * K_ + k], accC[nt][v]);
      atomicAdd(&xsb[(size_t)c * K_ + k], accS[nt][v]);
    }
  }
}

// x0[b,c] = sum_x h[b,c,x] * nw[b,x]
__global__ void k_x0(const f16* __restrict__ hA, const float* __restrict__ nw,
                     float* __restrict__ x0) {
  int b = blockIdx.y, c = blockIdx.x;
  __shared__ float red[256];
  float s = 0.f;
  const f16* row = hA + ((size_t)b * C_ + c) * N_;
  const float* nwb = nw + (size_t)b * N_;
  for (int x = threadIdx.x; x < N_; x += 256) s += (float)row[x] * nwb[x];
  red[threadIdx.x] = s;
  __syncthreads();
  for (int st = 128; st > 0; st >>= 1) {
    if (threadIdx.x < st) red[threadIdx.x] += red[threadIdx.x + st];
    __syncthreads();
  }
  if (threadIdx.x == 0) x0[(size_t)b * C_ + c] = red[0];
}

// Per-mode C x C mix; writes first 256 columns of the stacked weight matrix.
// Note: reference xs = -(h . wbs); our buffer holds +(h . wbs), so negate.
__global__ void k_modemix(const float* __restrict__ xc, const float* __restrict__ xs,
                          const float* __restrict__ Wc, const float* __restrict__ Ws,
                          f16* __restrict__ wstk) {
  int b = blockIdx.y, k = blockIdx.x, o = threadIdx.x;
  __shared__ float sc[C_], ss[C_];
  sc[o] = xc[((size_t)b * C_ + o) * K_ + k];
  ss[o] = -xs[((size_t)b * C_ + o) * K_ + k];
  __syncthreads();
  float fc = 0.f, fs = 0.f;
  for (int i = 0; i < C_; ++i) {
    float wc = Wc[((size_t)i * C_ + o) * K_ + k];
    float ws = Ws[((size_t)i * C_ + o) * K_ + k];
    fc += sc[i] * wc - ss[i] * ws;
    fs += ss[i] * wc + sc[i] * ws;
  }
  f16* row = wstk + ((size_t)b * C_ + o) * RS;
  row[k]      = (f16)(2.0f * fc);
  row[K_ + k] = (f16)(-2.0f * fs);
}

// Pack static columns (ws_w, gws_w) + bias = f0 + ws_b + gws_b.
__global__ void k_pack(const float* __restrict__ x0, const float* __restrict__ W0,
                       const float* __restrict__ wsw, const float* __restrict__ wsb,
                       const float* __restrict__ gww, const float* __restrict__ gwb,
                       f16* __restrict__ wstk, float* __restrict__ bias) {
  int b = blockIdx.y, o = blockIdx.x, j = threadIdx.x;  // 128 threads
  f16* row = wstk + ((size_t)b * C_ + o) * RS;
  row[2 * K_ + j]           = (f16)wsw[(size_t)o * C_ + j];
  row[3 * K_ + j]           = (f16)gww[(size_t)o * 3 * C_ + j];
  row[3 * K_ + C_ + j]      = (f16)gww[(size_t)o * 3 * C_ + C_ + j];
  row[3 * K_ + 2 * C_ + j]  = (f16)gww[(size_t)o * 3 * C_ + 2 * C_ + j];
  __shared__ float red[C_];
  red[j] = x0[(size_t)b * C_ + j] * W0[(size_t)j * C_ + o];
  __syncthreads();
  for (int st = 64; st > 0; st >>= 1) {
    if (j < st) red[j] += red[j + st];
    __syncthreads();
  }
  if (j == 0) bias[(size_t)b * C_ + o] = red[0] + wsb[o] + gwb[o];
}

// Graph gradient scatter: grad[b, src, d*128+c] += egw[b,e,d]*(h[dst,c]-h[src,c])
__global__ void k_edge(const float* __restrict__ hN, const int* __restrict__ edges,
                       const float* __restrict__ egw, float* __restrict__ grad) {
  int b = blockIdx.y, e = blockIdx.x, c = threadIdx.x;
  size_t be = (size_t)b * E_ + e;
  int s = edges[be * 2 + 0];
  int d = edges[be * 2 + 1];
  float diff = hN[((size_t)b * N_ + d) * C_ + c] - hN[((size_t)b * N_ + s) * C_ + c];
  float w0 = egw[be * 3 + 0], w1 = egw[be * 3 + 1], w2 = egw[be * 3 + 2];
  float* grow = grad + ((size_t)b * N_ + s) * (3 * C_);
  atomicAdd(&grow[c], w0 * diff);
  atomicAdd(&grow[C_ + c], w1 * diff);
  atomicAdd(&grow[2 * C_ + c], w2 * diff);
}

// g -> g/(1+|g|), as f16 rows 384..767 of the stacked B matrix.
__global__ void k_squash(const float* __restrict__ grad, f16* __restrict__ bstack) {
  size_t n  = (size_t)B_ * N_ * 3 * C_;
  size_t i  = (size_t)blockIdx.x * blockDim.x + threadIdx.x;
  size_t st = (size_t)gridDim.x * blockDim.x;
  for (; i < n; i += st) {
    float g = grad[i];
    float v = g / (1.0f + fabsf(g));
    size_t nx = i / (3 * C_);
    size_t j  = i % (3 * C_);
    bstack[nx * RS + 3 * K_ + j] = (f16)v;
  }
}

// Fused layer GEMM: hnew(128 x 16nodes) = Wstack(128x768) @ Bstack(768x16) + bias,
// epilogue exact-GELU (non-last) and re-emit h in all layouts. Register
// double-buffered so next step's loads overlap the current WMMA.
__global__ void __launch_bounds__(256)
k_fused(const f16* __restrict__ wstk, const f16* __restrict__ bstack,
        const float* __restrict__ bias, float* __restrict__ hN,
        f16* __restrict__ hA, f16* __restrict__ bstack_w, int last) {
  int b = blockIdx.y;
  int x0n = blockIdx.x * 16;
  int lane = threadIdx.x & 31, wv = threadIdx.x >> 5;
  int hi = lane >> 4, lo = lane & 15;
  int c0 = wv * 16;
  v8f acc = {0.f, 0.f, 0.f, 0.f, 0.f, 0.f, 0.f, 0.f};
  const f16* arow = wstk + ((size_t)b * C_ + c0 + lo) * RS;
  const f16* brow = bstack + ((size_t)b * N_ + x0n + lo) * RS;

  constexpr int NSTEP = RS / 32;  // 24
  v16h a_cur = ld2x8(arow + 8 * hi, arow + 16 + 8 * hi);
  v16h b_cur = ld2x8(brow + 8 * hi, brow + 16 + 8 * hi);
#pragma unroll
  for (int ks = 0; ks < NSTEP; ++ks) {
    v16h a_nxt = a_cur, b_nxt = b_cur;
    if (ks + 1 < NSTEP) {
      int r0 = (ks + 1) * 32;
      a_nxt = ld2x8(arow + r0 + 8 * hi, arow + r0 + 16 + 8 * hi);
      b_nxt = ld2x8(brow + r0 + 8 * hi, brow + r0 + 16 + 8 * hi);
    }
    acc = wmma16(a_cur, b_cur, acc);
    a_cur = a_nxt;
    b_cur = b_nxt;
  }
  // All waves in this block read rows 256..383 of the same 16 columns above;
  // ensure every wave is done before we overwrite those rows with the new h.
  __syncthreads();
  int x = x0n + lo;
  size_t nx = (size_t)b * N_ + x;
  v8h hpack;
#pragma unroll
  for (int v = 0; v < 8; ++v) {
    int c = c0 + 8 * hi + v;
    float fv = acc[v] + bias[(size_t)b * C_ + c];
    if (!last) fv = gelu_exact(fv);
    hN[nx * C_ + c] = fv;
    hpack[v] = (f16)fv;
    if (!last) hA[((size_t)b * C_ + c) * N_ + x] = (f16)fv;
  }
  if (!last) *(v8h*)(bstack_w + nx * RS + 2 * K_ + c0 + 8 * hi) = hpack;
}

// fc1 -> gelu -> fc2 -> residual
__global__ void k_final(const float* __restrict__ hN, const float* __restrict__ fc1w,
                        const float* __restrict__ fc1b, const float* __restrict__ fc2w,
                        const float* __restrict__ fc2b, const float* __restrict__ xin,
                        const float* __restrict__ tin, float* __restrict__ out) {
  int b = blockIdx.y, x = blockIdx.x, o = threadIdx.x;
  __shared__ float hrow[C_];
  __shared__ float z[FC_];
  size_t nx = (size_t)b * N_ + x;
  hrow[o] = hN[nx * C_ + o];
  __syncthreads();
  float s = fc1b[o];
  for (int c = 0; c < C_; ++c) s += hrow[c] * fc1w[(size_t)c * FC_ + o];
  z[o] = gelu_exact(s);
  __syncthreads();
  if (o < 3) {
    float acc = fc2b[o];
    for (int j = 0; j < FC_; ++j) acc += z[j] * fc2w[(size_t)j * 3 + o];
    out[nx * 3 + o] = xin[nx * 3 + o] + tin[nx] * acc;
  }
}

// ---------------- host orchestration ----------------
extern "C" void kernel_launch(void* const* d_in, const int* in_sizes, int n_in,
                              void* d_out, int out_size, void* d_ws, size_t ws_size,
                              hipStream_t stream) {
  (void)in_sizes; (void)n_in; (void)out_size; (void)ws_size;
  const float* xin  = (const float*)d_in[0];
  const float* tin  = (const float*)d_in[1];
  // d_in[2] node_mask: unused (reference deletes it)
  const float* nodes = (const float*)d_in[3];
  const float* nodw  = (const float*)d_in[4];
  const int*   edges = (const int*)d_in[5];
  const float* egw   = (const float*)d_in[6];
  const float* modes = (const float*)d_in[7];
  const float* spL   = (const float*)d_in[8];
  const float* fc0w  = (const float*)d_in[9];
  const float* fc0b  = (const float*)d_in[10];
  const float* Wc    = (const float*)d_in[11];
  const float* Ws    = (const float*)d_in[12];
  const float* W0    = (const float*)d_in[13];
  const float* wsw   = (const float*)d_in[14];
  const float* wsb   = (const float*)d_in[15];
  const float* gww   = (const float*)d_in[16];
  const float* gwb   = (const float*)d_in[17];
  const float* fc1w  = (const float*)d_in[18];
  const float* fc1b  = (const float*)d_in[19];
  const float* fc2w  = (const float*)d_in[20];
  const float* fc2b  = (const float*)d_in[21];

  char* ws = (char*)d_ws;
  size_t off = 0;
  auto take = [&](size_t sz) {
    size_t o = off;
    off = (off + sz + 255) & ~(size_t)255;
    return o;
  };
  f16*   bstack = (f16*)(ws + take((size_t)B_ * N_ * RS * sizeof(f16)));
  f16*   hA     = (f16*)(ws + take((size_t)B_ * C_ * N_ * sizeof(f16)));
  float* hN     = (float*)(ws + take((size_t)B_ * N_ * C_ * sizeof(float)));
  f16*   wb     = (f16*)(ws + take((size_t)B_ * 2 * K_ * N_ * sizeof(f16)));
  float* nw     = (float*)(ws + take((size_t)B_ * N_ * sizeof(float)));
  float* xc     = (float*)(ws + take((size_t)B_ * C_ * K_ * sizeof(float)));
  float* xs     = (float*)(ws + take((size_t)B_ * C_ * K_ * sizeof(float)));
  float* x0     = (float*)(ws + take((size_t)B_ * C_ * sizeof(float)));
  f16*   wstk   = (f16*)(ws + take((size_t)B_ * C_ * RS * sizeof(f16)));
  float* bias   = (float*)(ws + take((size_t)B_ * C_ * sizeof(float)));
  float* grad   = (float*)(ws + take((size_t)B_ * N_ * 3 * C_ * sizeof(float)));

  dim3 nodeGrid((N_ + 63) / 64, B_);
  k_basis<<<nodeGrid, 64, 0, stream>>>(nodes, nodw, modes, spL, bstack, wb, nw);
  k_fc0<<<nodeGrid, 64, 0, stream>>>(xin, tin, fc0w, fc0b, hN, hA, bstack);

  constexpr size_t CCK = (size_t)C_ * C_ * K_;
  for (int l = 0; l < L_; ++l) {
    k_zero<<<256, 256, 0, stream>>>(xc, (size_t)B_ * C_ * K_);
    k_zero<<<256, 256, 0, stream>>>(xs, (size_t)B_ * C_ * K_);
    k_zero<<<4, 256, 0, stream>>>(x0, (size_t)B_ * C_);
    k_zero<<<2048, 256, 0, stream>>>(grad, (size_t)B_ * N_ * 3 * C_);

    k_fwd<<<dim3(FWD_CHUNKS, B_), 256, 0, stream>>>(hA, wb, xc, xs);
    k_x0<<<dim3(C_, B_), 256, 0, stream>>>(hA, nw, x0);
    k_modemix<<<dim3(K_, B_), 128, 0, stream>>>(xc, xs, Wc + (size_t)l * CCK,
                                                Ws + (size_t)l * CCK, wstk);
    k_pack<<<dim3(C_, B_), 128, 0, stream>>>(
        x0, W0 + (size_t)l * C_ * C_, wsw + (size_t)l * C_ * C_,
        wsb + (size_t)l * C_, gww + (size_t)l * C_ * 3 * C_,
        gwb + (size_t)l * C_, wstk, bias);
    k_edge<<<dim3(E_, B_), 128, 0, stream>>>(hN, edges, egw, grad);
    k_squash<<<4096, 256, 0, stream>>>(grad, bstack);
    k_fused<<<dim3(N_ / 16, B_), 256, 0, stream>>>(wstk, bstack, bias, hN, hA,
                                                   bstack, (l == L_ - 1) ? 1 : 0);
  }

  k_final<<<dim3(N_, B_), 128, 0, stream>>>(hN, fc1w, fc1b, fc2w, fc2b, xin, tin,
                                            (float*)d_out);
}
